// DiscreteLatentModelGroups_19705309954240
// MI455X (gfx1250) — matrix operand order repaired and checked by reading, not compiled, not run on hardware
//
#include <hip/hip_runtime.h>

// ---------------------------------------------------------------------------
// CDNA5 (gfx1250) WMMA types
// ---------------------------------------------------------------------------
typedef _Float16 v16h __attribute__((ext_vector_type(16)));
typedef _Float16 v8h  __attribute__((ext_vector_type(8)));
typedef float    v8f  __attribute__((ext_vector_type(8)));

#define CF_RIN  1   // relu applied to input activations on load
#define CF_ROUT 2   // relu applied to output
#define CF_ADD  4   // out += Add[same index]
#define CF_BIAS 8   // add bias

#define KMAX 1152   // largest Cin_pg*kh*kw in the model (128*3*3), mult of 32

struct ConvP {
    int N;
    int Cin_total, in_c_off, Cin;   // input tensor full channels; slice used
    int Hin, Win;
    int Cout_total, out_c_off, Cout;
    int Hout, Wout;
    int stride, pad;
    int groups;
    int flags;
};

// assemble a v16h WMMA fragment from two contiguous 8-half runs
static __device__ inline v16h frag16(v8h lo, v8h hi2)
{
    v16h a;
    #pragma unroll
    for (int e = 0; e < 8; ++e) { a[e] = lo[e]; a[8 + e] = hi2[e]; }
    return a;
}

// ---------------------------------------------------------------------------
// Grouped conv (incl. transposed conv via compile-time lhs dilation LDIL) as
// implicit GEMM on v_wmma_f32_16x16x32_f16.
//   A = weight tile (16 cout rows x K), staged once in LDS as f16; fragments
//       are two aligned 16B ds loads per k-chunk.
//   B = im2col input (K x 16 pixels) gathered per lane (true gather).
// 256 threads = 8 waves/block share one LDS weight tile; each wave computes a
// different 16-pixel output tile.  KH/KW compile-time -> constant-divisor ops.
// ---------------------------------------------------------------------------
template <int KH, int KW, int LDIL>
__global__ __launch_bounds__(256) void k_conv_wmma(
    const float* __restrict__ X, const float* __restrict__ Wt,
    const float* __restrict__ Bias, const float* __restrict__ Add,
    float* __restrict__ Out, ConvP p)
{
    __shared__ __align__(16) _Float16 aL[16 * KMAX];   // 36 KB of 320 KB/WGP

    const int tid  = threadIdx.x;
    const int lane = tid & 31;
    const int wv   = tid >> 5;
    const int l16  = lane & 15;
    const int hi   = lane >> 4;

    const int Cout_pg = p.Cout / p.groups;
    const int Cin_pg  = p.Cin  / p.groups;
    const int mtiles  = (Cout_pg + 15) >> 4;
    const int g       = blockIdx.x / mtiles;
    const int mt      = blockIdx.x % mtiles;

    constexpr int KHW = KH * KW;
    const int Ktot = Cin_pg * KHW;
    const int K32  = (Ktot + 31) & ~31;     // zero-padded K extent in LDS

    // ---- cooperative weight-tile stage: rows = cout within tile, coalesced k
    for (int row = 0; row < 16; ++row) {
        const int m = mt * 16 + row;
        for (int k = tid; k < K32; k += 256) {
            float v = 0.f;
            if (m < Cout_pg && k < Ktot)
                v = Wt[(size_t)(g * Cout_pg + m) * Ktot + k];
            aL[row * K32 + k] = (_Float16)v;
        }
    }
    __syncthreads();

    const int npix = p.N * p.Hout * p.Wout;
    const int p0   = (blockIdx.y * 8 + wv) * 16;
    if (p0 >= npix) return;                 // wave-uniform; after the only barrier

    const int  pix    = p0 + l16;
    const bool pvalid = pix < npix;
    const int  pc     = pvalid ? pix : 0;
    const int  b_     = pc / (p.Hout * p.Wout);
    const int  rem    = pc % (p.Hout * p.Wout);
    const int  oy     = rem / p.Wout;
    const int  ox     = rem % p.Wout;
    const int  iyb    = oy * p.stride - p.pad;   // hoisted gather bases
    const int  ixb    = ox * p.stride - p.pad;

    const bool relu_in = (p.flags & CF_RIN) != 0;
    const size_t xbase = ((size_t)b_ * p.Cin_total + p.in_c_off + g * Cin_pg)
                         * p.Hin * p.Win;

    const _Float16* arow = aL + l16 * K32 + 8 * hi;  // lane's A runs

    v8f acc = {0.f,0.f,0.f,0.f,0.f,0.f,0.f,0.f};

    for (int kb = 0; kb < K32; kb += 32) {
        // ---- A fragment: two aligned 16B LDS vector loads
        const v8h alo = *(const v8h*)(arow + kb);
        const v8h ahi = *(const v8h*)(arow + kb + 16);
        const v16h a  = frag16(alo, ahi);

        // ---- B fragment: im2col gather (32x16 f16 layout, half selects +8K)
        v16h b;
        #pragma unroll
        for (int j = 0; j < 8; ++j) {
            const int kb0 = 2*j + 8*hi + 8*(j >> 2);
            #pragma unroll
            for (int e = 0; e < 2; ++e) {
                const int k = kb + kb0 + e;
                float v = 0.f;
                if (pvalid && k < Ktot) {
                    const int ci = k / KHW;
                    const int r  = k % KHW;
                    const int iyn = iyb + r / KW;
                    const int ixn = ixb + r % KW;
                    int iy, ix; bool ok;
                    if (LDIL > 1) {
                        ok = (iyn >= 0) && (ixn >= 0) &&
                             ((iyn & (LDIL - 1)) == 0) && ((ixn & (LDIL - 1)) == 0);
                        iy = iyn / LDIL; ix = ixn / LDIL;
                    } else { ok = true; iy = iyn; ix = ixn; }
                    ok = ok && iy >= 0 && iy < p.Hin && ix >= 0 && ix < p.Win;
                    if (ok) {
                        v = X[xbase + ((size_t)ci * p.Hin + iy) * p.Win + ix];
                        if (relu_in && v < 0.f) v = 0.f;
                    }
                }
                b[2*j + e] = (_Float16)v;
            }
        }
        acc = __builtin_amdgcn_wmma_f32_16x16x32_f16(
                  false, a, false, b, (short)0, acc, false, false);
    }

    if (!pvalid) return;

    #pragma unroll
    for (int j = 0; j < 8; ++j) {
        const int m = mt * 16 + j + 8*hi;   // C/D layout: row = j (+8 high half)
        if (m < Cout_pg) {
            float v = acc[j];
            if (p.flags & CF_BIAS) v += Bias[g * Cout_pg + m];
            const size_t oidx = (((size_t)b_ * p.Cout_total + p.out_c_off + g*Cout_pg + m)
                                 * p.Hout + oy) * p.Wout + ox;
            if (p.flags & CF_ADD)  v += Add[oidx];
            if ((p.flags & CF_ROUT) && v < 0.f) v = 0.f;
            Out[oidx] = v;
        }
    }
}

// ---------------------------------------------------------------------------
// VQ: argmin over Kc codes of |z|^2+|e|^2-2 z.e, dot products via WMMA.
// Block = 128 threads / 4 waves / 64 pixels.  The 64x128 z panel is staged in
// LDS as f16 in [pixel][feature] order (stride 136 halfs -> bank step 4*px),
// so each WMMA B fragment is two aligned 16B ds loads.  A fragments are two
// 16B global loads from the pre-converted f16 codebook.  Lane-halves see
// different 8-code rows per 16-code tile; combined via __shfl_xor(16).
// ---------------------------------------------------------------------------
#define ZP 136   // padded feature stride (128 + 8 halfs)

__global__ __launch_bounds__(128) void k_vq_wmma(
    const float* __restrict__ Z, const float* __restrict__ Emb,
    const _Float16* __restrict__ EmbH, const float* __restrict__ Enorm,
    float* __restrict__ Q, float* __restrict__ Lbuf,
    int Npix, int Cc, int Kc, int HW)
{
    __shared__ __align__(16) _Float16 zL[64 * ZP];   // 17 KB

    const int tid  = threadIdx.x;
    const int lane = tid & 31;
    const int wv   = tid >> 5;
    const int l16  = lane & 15;
    const int hi   = lane >> 4;

    const int pb = blockIdx.x * 64;         // block pixel base

    // ---- cooperative z-panel stage (coalesced over pixels within a feature)
    for (int idx = tid; idx < Cc * 64; idx += 128) {
        const int px = idx & 63;
        const int c  = idx >> 6;
        const int pg = pb + px;
        float v = 0.f;
        if (pg < Npix)
            v = Z[((size_t)(pg / HW) * Cc + c) * HW + (pg % HW)];
        zL[px * ZP + c] = (_Float16)v;
    }
    __syncthreads();

    const int px = wv * 16 + l16;           // this lane's pixel within block
    const int  pix    = pb + px;
    const bool pvalid = pix < Npix;
    const int  pc     = pvalid ? pix : 0;
    const int  b_     = pc / HW;
    const int  rem    = pc % HW;

    // |z|^2 for this lane's pixel (from the staged f16 panel)
    const _Float16* zrow = zL + px * ZP;
    float zn = 0.f;
    for (int c = 0; c < Cc; ++c) {
        const float v = (float)zrow[c];
        zn += v * v;
    }

    const _Float16* brun = zrow + 8 * hi;   // lane's B run base (runs at +0, +16)

    float bestd = 3.402823e38f;
    int   besti = 0;

    for (int t = 0; t < (Kc >> 4); ++t) {
        v8f acc = {0.f,0.f,0.f,0.f,0.f,0.f,0.f,0.f};
        const _Float16* arun = EmbH + (size_t)(t * 16 + l16) * Cc + 8 * hi;
        for (int kb = 0; kb < Cc; kb += 32) {
            // A: two 16B global vector loads from f16 codebook row
            const v16h a = frag16(*(const v8h*)(arun + kb),
                                  *(const v8h*)(arun + kb + 16));
            // B: two 16B LDS vector loads (contiguous features)
            const v16h b = frag16(*(const v8h*)(brun + kb),
                                  *(const v8h*)(brun + kb + 16));
            acc = __builtin_amdgcn_wmma_f32_16x16x32_f16(
                      false, a, false, b, (short)0, acc, false, false);
        }
        #pragma unroll
        for (int j = 0; j < 8; ++j) {
            const int code = t * 16 + j + 8*hi;
            const float d  = zn + Enorm[code] - 2.f * acc[j];
            if (d < bestd) { bestd = d; besti = code; }
        }
    }

    // combine the two lane-halves (each saw half of the codes for this pixel)
    const float od = __shfl_xor(bestd, 16, 32);
    const int   oi = __shfl_xor(besti, 16, 32);
    if (od < bestd || (od == bestd && oi < besti)) { bestd = od; besti = oi; }

    if (!pvalid) return;                    // partner lane has same pvalid

    // write q (straight-through == q, exact f32 codebook) and per-pixel sq err
    float lsum = 0.f;
    const int c0 = hi * (Cc >> 1);
    for (int c = c0; c < c0 + (Cc >> 1); ++c) {
        const float  e  = Emb[(size_t)besti * Cc + c];
        const size_t zi = ((size_t)b_ * Cc + c) * HW + rem;
        const float  zv = Z[zi];
        Q[zi] = e;
        const float dd = e - zv;
        lsum += dd * dd;
    }
    lsum += __shfl_xor(lsum, 16, 32);
    if (hi == 0) Lbuf[pix] = lsum;
}

// ---------------------------------------------------------------------------
// Small helpers
// ---------------------------------------------------------------------------
// codebook prep: squared norms + f16 copy for WMMA A-fragments
__global__ void k_emb_prep(const float* __restrict__ E, float* __restrict__ En,
                           _Float16* __restrict__ Eh, int K, int C)
{
    const int i = blockIdx.x * blockDim.x + threadIdx.x;
    if (i >= K) return;
    float s = 0.f;
    for (int c = 0; c < C; ++c) {
        const float v = E[(size_t)i * C + c];
        Eh[(size_t)i * C + c] = (_Float16)v;
        s += v * v;
    }
    En[i] = s;
}

// torch ConvTranspose2d weight (in_ch, out_pg, kh, kw) -> flipped OIHW
// (groups*out_pg, in_pg, kh, kw) for the lhs-dilated conv formulation.
__global__ void k_prep_tw(const float* __restrict__ Ws, float* __restrict__ Wd,
                          int in_ch, int outg, int kh, int kw, int groups)
{
    const int total = in_ch * outg * kh * kw;
    const int i = blockIdx.x * blockDim.x + threadIdx.x;
    if (i >= total) return;
    const int in_pg = in_ch / groups;
    const int kxy = i % (kh * kw);
    int t = i / (kh * kw);
    const int ipg = t % in_pg; t /= in_pg;
    const int o   = t % outg;
    const int g   = t / outg;
    const int ky = kxy / kw, kx = kxy % kw;
    Wd[i] = Ws[(((size_t)(g * in_pg + ipg) * outg + o) * kh + (kh - 1 - ky)) * kw
               + (kw - 1 - kx)];
}

__global__ void k_copy_ch(const float* __restrict__ S, float* __restrict__ D,
                          int N, int HW, int Cs, int cs, int cn, int Cd, int ds)
{
    const size_t total = (size_t)N * cn * HW;
    const size_t i = (size_t)blockIdx.x * blockDim.x + threadIdx.x;
    if (i >= total) return;
    const int hw = (int)(i % HW);
    size_t t = i / HW;
    const int c = (int)(t % cn);
    const int n = (int)(t / cn);
    D[((size_t)n * Cd + ds + c) * HW + hw] = S[((size_t)n * Cs + cs + c) * HW + hw];
}

// deterministic fixed-order reduction -> loss scalar
__global__ void k_reduce(const float* __restrict__ v, int n, float scale,
                         float* __restrict__ out)
{
    __shared__ float sm[256];
    float s = 0.f;
    for (int i = threadIdx.x; i < n; i += 256) s += v[i];
    sm[threadIdx.x] = s;
    __syncthreads();
    for (int st = 128; st > 0; st >>= 1) {
        if ((int)threadIdx.x < st) sm[threadIdx.x] += sm[threadIdx.x + st];
        __syncthreads();
    }
    if (threadIdx.x == 0) out[0] = sm[0] * scale;
}

// ---------------------------------------------------------------------------
// Host-side helpers
// ---------------------------------------------------------------------------
static inline ConvP mk(int N, int CinT, int inOff, int Cin, int Hin, int Win,
                       int CoutT, int outOff, int Cout, int Hout, int Wout,
                       int st, int pad, int g, int fl)
{
    ConvP p; p.N = N; p.Cin_total = CinT; p.in_c_off = inOff; p.Cin = Cin;
    p.Hin = Hin; p.Win = Win; p.Cout_total = CoutT; p.out_c_off = outOff;
    p.Cout = Cout; p.Hout = Hout; p.Wout = Wout;
    p.stride = st; p.pad = pad; p.groups = g; p.flags = fl;
    return p;
}

static void conv_go(hipStream_t s, const float* X, const float* Wt,
                    const float* Bias, const float* Add, float* Out,
                    ConvP p, int kh, int ldil)
{
    const int Cout_pg = p.Cout / p.groups;
    const int mtiles  = (Cout_pg + 15) / 16;
    const int npix    = p.N * p.Hout * p.Wout;
    const int ptiles  = (npix + 15) / 16;
    dim3 grid(p.groups * mtiles, (ptiles + 7) / 8);
    if (kh == 1)
        k_conv_wmma<1,1,1><<<grid, 256, 0, s>>>(X, Wt, Bias, Add, Out, p);
    else if (kh == 3)
        k_conv_wmma<3,3,1><<<grid, 256, 0, s>>>(X, Wt, Bias, Add, Out, p);
    else if (ldil == 1)
        k_conv_wmma<4,4,1><<<grid, 256, 0, s>>>(X, Wt, Bias, Add, Out, p);
    else
        k_conv_wmma<4,4,2><<<grid, 256, 0, s>>>(X, Wt, Bias, Add, Out, p);
}

// res_stack: for each layer: h=conv3x3(relu(x)) g2 -> rbuf(64ch);
// x = x + conv1x1(relu(rbuf)) g2; relu at the very end.
static void res_go(hipStream_t s, float* xbuf, const float* w1, const float* w2,
                   float* rbuf, int N, int Hs, int L)
{
    for (int i = 0; i < L; ++i) {
        conv_go(s, xbuf, w1 + (size_t)i * 64 * 64 * 9, nullptr, nullptr, rbuf,
                mk(N, 128, 0, 128, Hs, Hs, 64, 0, 64, Hs, Hs, 1, 1, 2, CF_RIN),
                3, 1);
        const int fl = CF_RIN | CF_ADD | ((i == L - 1) ? CF_ROUT : 0);
        conv_go(s, rbuf, w2 + (size_t)i * 128 * 32, nullptr, xbuf, xbuf,
                mk(N, 64, 0, 64, Hs, Hs, 128, 0, 128, Hs, Hs, 1, 0, 2, fl),
                1, 1);
    }
}

// ---------------------------------------------------------------------------
extern "C" void kernel_launch(void* const* d_in, const int* in_sizes, int n_in,
                              void* d_out_, int out_size, void* d_ws, size_t ws_size,
                              hipStream_t stream)
{
    (void)in_sizes; (void)n_in; (void)out_size; (void)ws_size;

    const float* x      = (const float*)d_in[0];
    const float* eb_dw  = (const float*)d_in[1];
    const float* eb_db  = (const float*)d_in[2];
    const float* eb_rw  = (const float*)d_in[3];
    const float* eb_rb  = (const float*)d_in[4];
    const float* eb_c1w = (const float*)d_in[5];
    const float* eb_c1b = (const float*)d_in[6];
    const float* eb_c2w = (const float*)d_in[7];
    const float* eb_c2b = (const float*)d_in[8];
    const float* eb_c3w = (const float*)d_in[9];
    const float* eb_c3b = (const float*)d_in[10];
    const float* eb_rw1 = (const float*)d_in[11];
    const float* eb_rw2 = (const float*)d_in[12];
    const float* et_c1w = (const float*)d_in[13];
    const float* et_c1b = (const float*)d_in[14];
    const float* et_c2w = (const float*)d_in[15];
    const float* et_c2b = (const float*)d_in[16];
    const float* et_rw1 = (const float*)d_in[17];
    const float* et_rw2 = (const float*)d_in[18];
    const float* pvt_w  = (const float*)d_in[19];
    const float* pvt_b  = (const float*)d_in[20];
    const float* up_w   = (const float*)d_in[21];
    const float* up_b   = (const float*)d_in[22];
    const float* pvb_w  = (const float*)d_in[23];
    const float* pvb_b  = (const float*)d_in[24];
    const float* emb_t  = (const float*)d_in[25];
    const float* emb_b  = (const float*)d_in[26];
    const float* db_c1w = (const float*)d_in[27];
    const float* db_c1b = (const float*)d_in[28];
    const float* db_rw1 = (const float*)d_in[29];
    const float* db_rw2 = (const float*)d_in[30];
    const float* db_t1w = (const float*)d_in[31];
    const float* db_t1b = (const float*)d_in[32];
    const float* db_t2w = (const float*)d_in[33];
    const float* db_t2b = (const float*)d_in[34];
    const float* db_odw = (const float*)d_in[35];
    const float* db_odb = (const float*)d_in[36];
    const float* db_orw = (const float*)d_in[37];
    const float* db_orb = (const float*)d_in[38];

    float* out = (float*)d_out_;
    // output layout: [loss_b, loss_t, recon(8*4*256*256), q_t(8*128*32*32), q_b(8*128*64*64)]
    float* loss_b = out + 0;
    float* loss_t = out + 1;
    float* recon  = out + 2;
    float* q_t    = recon + (size_t)8 * 4 * 256 * 256;
    float* q_b    = q_t   + (size_t)8 * 128 * 32 * 32;

    // ---- workspace bump layout (floats) ----
    float* ws = (float*)d_ws;
    size_t o = 0;
    float* h0    = ws + o; o += (size_t)8 * 64 * 256 * 256;   // also reused joined/dj
    float* h1    = ws + o; o += (size_t)8 * 64 * 128 * 128;
    float* h2    = ws + o; o += (size_t)8 * 128 * 64 * 64;
    float* enc_b = ws + o; o += (size_t)8 * 128 * 64 * 64;
    float* rbuf  = ws + o; o += (size_t)8 * 64 * 64 * 64;
    float* t1    = ws + o; o += (size_t)8 * 128 * 32 * 32;
    float* enc_t = ws + o; o += (size_t)8 * 128 * 32 * 32;
    float* zt    = ws + o; o += (size_t)8 * 128 * 32 * 32;
    float* up_t  = ws + o; o += (size_t)8 * 128 * 64 * 64;
    float* zb    = ws + o; o += (size_t)8 * 128 * 64 * 64;
    float* y1    = ws + o; o += (size_t)8 * 128 * 64 * 64;
    float* y2    = ws + o; o += (size_t)8 * 64 * 128 * 128;
    float* y3    = ws + o; o += (size_t)8 * 32 * 256 * 256;
    float* wf_up = ws + o; o += (size_t)128 * 64 * 16;
    float* wf_t1 = ws + o; o += (size_t)64 * 64 * 16;
    float* wf_t2 = ws + o; o += (size_t)32 * 32 * 16;
    float* en_t  = ws + o; o += 2048;
    float* en_b  = ws + o; o += 2048;
    float* ls_t  = ws + o; o += 8192;
    float* ls_b  = ws + o; o += 32768;
    _Float16* eh_t = (_Float16*)(ws + o); o += (size_t)2048 * 128 / 2;  // f16 codebook
    _Float16* eh_b = (_Float16*)(ws + o); o += (size_t)2048 * 128 / 2;
    float* joined = h0;                       // h0 dead after h1 is built
    float* dj     = h0 + (size_t)8 * 256 * 64 * 64;

    // ================= EncoderBot =================
    conv_go(stream, x, eb_dw, eb_db, nullptr, h0,
            mk(8, 4, 0, 1, 256, 256, 64, 0, 32, 256, 256, 1, 0, 1, CF_ROUT|CF_BIAS), 1, 1);
    conv_go(stream, x, eb_rw, eb_rb, nullptr, h0,
            mk(8, 4, 1, 3, 256, 256, 64, 32, 32, 256, 256, 1, 0, 1, CF_ROUT|CF_BIAS), 1, 1);
    conv_go(stream, h0, eb_c1w, eb_c1b, nullptr, h1,
            mk(8, 64, 0, 64, 256, 256, 64, 0, 64, 128, 128, 2, 1, 2, CF_ROUT|CF_BIAS), 4, 1);
    conv_go(stream, h1, eb_c2w, eb_c2b, nullptr, h2,
            mk(8, 64, 0, 64, 128, 128, 128, 0, 128, 64, 64, 2, 1, 2, CF_ROUT|CF_BIAS), 4, 1);
    conv_go(stream, h2, eb_c3w, eb_c3b, nullptr, enc_b,
            mk(8, 128, 0, 128, 64, 64, 128, 0, 128, 64, 64, 1, 1, 2, CF_BIAS), 3, 1);
    res_go(stream, enc_b, eb_rw1, eb_rw2, rbuf, 8, 64, 2);

    // ================= EncoderTop =================
    conv_go(stream, enc_b, et_c1w, et_c1b, nullptr, t1,
            mk(8, 128, 0, 128, 64, 64, 128, 0, 128, 32, 32, 2, 1, 2, CF_ROUT|CF_BIAS), 4, 1);
    conv_go(stream, t1, et_c2w, et_c2b, nullptr, enc_t,
            mk(8, 128, 0, 128, 32, 32, 128, 0, 128, 32, 32, 1, 1, 2, CF_ROUT|CF_BIAS), 3, 1);
    res_go(stream, enc_t, et_rw1, et_rw2, rbuf, 8, 32, 2);

    // ================= top VQ =================
    conv_go(stream, enc_t, pvt_w, pvt_b, nullptr, zt,
            mk(8, 128, 0, 128, 32, 32, 128, 0, 128, 32, 32, 1, 0, 2, CF_BIAS), 1, 1);
    k_emb_prep<<<2048/256, 256, 0, stream>>>(emb_t, en_t, eh_t, 2048, 128);
    {
        const int Npix = 8 * 32 * 32;                      // 8192
        k_vq_wmma<<<Npix/64, 128, 0, stream>>>(zt, emb_t, eh_t, en_t, q_t, ls_t,
                                               Npix, 128, 2048, 32*32);
        k_reduce<<<1, 256, 0, stream>>>(ls_t, Npix, 0.25f / ((float)Npix * 128.f), loss_t);
    }

    // up_t = conv_t(q_t): flip/regroup weights then lhs-dilated conv
    k_prep_tw<<<(128*64*16 + 255)/256, 256, 0, stream>>>(up_w, wf_up, 128, 64, 4, 4, 2);
    conv_go(stream, q_t, wf_up, up_b, nullptr, up_t,
            mk(8, 128, 0, 128, 32, 32, 128, 0, 128, 64, 64, 1, 2, 2, CF_BIAS), 4, 2);

    // ================= bottom VQ =================
    {
        const size_t tot = (size_t)8 * 64 * 4096;
        const int blk = (int)((tot + 255) / 256);
        k_copy_ch<<<blk, 256, 0, stream>>>(enc_b, joined, 8, 4096, 128, 0,  64, 256,   0);
        k_copy_ch<<<blk, 256, 0, stream>>>(up_t,  joined, 8, 4096, 128, 0,  64, 256,  64);
        k_copy_ch<<<blk, 256, 0, stream>>>(enc_b, joined, 8, 4096, 128, 64, 64, 256, 128);
        k_copy_ch<<<blk, 256, 0, stream>>>(up_t,  joined, 8, 4096, 128, 64, 64, 256, 192);
    }
    conv_go(stream, joined, pvb_w, pvb_b, nullptr, zb,
            mk(8, 256, 0, 256, 64, 64, 128, 0, 128, 64, 64, 1, 0, 2, CF_BIAS), 1, 1);
    k_emb_prep<<<2048/256, 256, 0, stream>>>(emb_b, en_b, eh_b, 2048, 128);
    {
        const int Npix = 8 * 64 * 64;                      // 32768
        k_vq_wmma<<<Npix/64, 128, 0, stream>>>(zb, emb_b, eh_b, en_b, q_b, ls_b,
                                               Npix, 128, 2048, 64*64);
        k_reduce<<<1, 256, 0, stream>>>(ls_b, Npix, 0.25f / ((float)Npix * 128.f), loss_b);
    }

    // ================= DecoderBot =================
    {
        const size_t tot = (size_t)8 * 64 * 4096;
        const int blk = (int)((tot + 255) / 256);
        k_copy_ch<<<blk, 256, 0, stream>>>(up_t, dj, 8, 4096, 128, 0,  64, 256,   0);
        k_copy_ch<<<blk, 256, 0, stream>>>(q_b,  dj, 8, 4096, 128, 0,  64, 256,  64);
        k_copy_ch<<<blk, 256, 0, stream>>>(up_t, dj, 8, 4096, 128, 64, 64, 256, 128);
        k_copy_ch<<<blk, 256, 0, stream>>>(q_b,  dj, 8, 4096, 128, 64, 64, 256, 192);
    }
    conv_go(stream, dj, db_c1w, db_c1b, nullptr, y1,
            mk(8, 256, 0, 256, 64, 64, 128, 0, 128, 64, 64, 1, 1, 2, CF_BIAS), 3, 1);
    res_go(stream, y1, db_rw1, db_rw2, rbuf, 8, 64, 2);

    k_prep_tw<<<(128*32*16 + 255)/256, 256, 0, stream>>>(db_t1w, wf_t1, 128, 32, 4, 4, 2);
    conv_go(stream, y1, wf_t1, db_t1b, nullptr, y2,
            mk(8, 128, 0, 128, 64, 64, 64, 0, 64, 128, 128, 1, 2, 2, CF_ROUT|CF_BIAS), 4, 2);

    k_prep_tw<<<(64*16*16 + 255)/256, 256, 0, stream>>>(db_t2w, wf_t2, 64, 16, 4, 4, 2);
    conv_go(stream, y2, wf_t2, db_t2b, nullptr, y3,
            mk(8, 64, 0, 64, 128, 128, 32, 0, 32, 256, 256, 1, 2, 2, CF_BIAS), 4, 2);

    // final 1x1 heads into recon (channel 0 = depth, 1..3 = rgb)
    conv_go(stream, y3, db_odw, db_odb, nullptr, recon,
            mk(8, 32, 0, 16, 256, 256, 4, 0, 1, 256, 256, 1, 0, 1, CF_BIAS), 1, 1);
    conv_go(stream, y3, db_orw, db_orb, nullptr, recon,
            mk(8, 32, 16, 16, 256, 256, 4, 1, 3, 256, 256, 1, 0, 1, CF_BIAS), 1, 1);
}
